// AttnBlock_11347303596463
// MI455X (gfx1250) — matrix operand order repaired and checked by reading, compiled, not run on hardware
//
#include <hip/hip_runtime.h>
#include <hip/hip_bf16.h>
#include <math.h>

// Problem sizes (fixed by reference)
#define CC   512
#define NN   2048
#define BBATCH 8
#define GG   32
#define EPSF 1e-6f
#define QSCALE 0.04419417382415922f   // 1/sqrt(512)

typedef __bf16 bh;
typedef __attribute__((ext_vector_type(16))) __bf16 v16bh;
typedef __attribute__((ext_vector_type(8)))  float  v8f;
typedef int int4v __attribute__((__vector_size__(4 * sizeof(int))));

#define BM 128
#define BN 128
#define BK 64     // two 16x16x32 K-steps per LDS stage
#define KPAD 72   // 144B row stride: 16B-aligned, conflict-free 16-lane fragment reads

// gfx1250 async global->LDS staging (ASYNCcnt path), with safe fallback.
#if defined(__has_builtin)
#if __has_builtin(__builtin_amdgcn_global_load_async_to_lds_b128) && \
    __has_builtin(__builtin_amdgcn_s_wait_asynccnt)
#define USE_ASYNC_LDS 1
#endif
#endif
#ifndef USE_ASYNC_LDS
#define USE_ASYNC_LDS 0
#endif

#define AS_GLOBAL __attribute__((address_space(1)))
#define AS_LDS    __attribute__((address_space(3)))

union FragBH { uint4 q[2]; v16bh v; };

// ---------------------------------------------------------------------------
// GEMM core: C[128x128 f32] += A[MxK bf16 row-major] x Bt[NxK bf16 row-major]^T
// 256 threads = 8 waves in a 2x4 (m,n) grid; each wave: 4x2 = 8 wmma tiles.
// ---------------------------------------------------------------------------
__device__ __forceinline__ void gemm_core(const bh* __restrict__ A, int lda,
                                          const bh* __restrict__ Bt, int ldb,
                                          int K, int mBlk, int nBlk,
                                          v8f (&acc)[4][2])
{
    __shared__ bh sA[2][BM][KPAD];
    __shared__ bh sB[2][BN][KPAD];

    const int tid  = threadIdx.x;
    const int lane = tid & 31;
    const int wv   = tid >> 5;
    const int wm   = (wv >> 2) * 64;   // wave row offset within block tile
    const int wn   = (wv & 3)  * 32;   // wave col offset within block tile
    const int lr   = tid >> 3;         // 0..31 (staging row)
    const int lc   = (tid & 7) * 8;    // 0..56 (staging col, 16B chunks)

    const bh* aBase = A  + (long)(mBlk + lr) * lda + lc;
    const bh* bBase = Bt + (long)(nBlk + lr) * ldb + lc;

    auto stage = [&](int kt, int buf) {
#pragma unroll
        for (int p = 0; p < 4; ++p) {
            const bh* ag = aBase + (long)kt * BK + (long)(p * 32) * lda;
            const bh* bg = bBase + (long)kt * BK + (long)(p * 32) * ldb;
#if USE_ASYNC_LDS
            __builtin_amdgcn_global_load_async_to_lds_b128(
                (AS_GLOBAL int4v*)(uintptr_t)ag,
                (AS_LDS int4v*)(unsigned int)(uintptr_t)&sA[buf][lr + p * 32][lc],
                0, 0);
            __builtin_amdgcn_global_load_async_to_lds_b128(
                (AS_GLOBAL int4v*)(uintptr_t)bg,
                (AS_LDS int4v*)(unsigned int)(uintptr_t)&sB[buf][lr + p * 32][lc],
                0, 0);
#else
            *(uint4*)&sA[buf][lr + p * 32][lc] = *(const uint4*)ag;
            *(uint4*)&sB[buf][lr + p * 32][lc] = *(const uint4*)bg;
#endif
        }
    };

    const v8f zero = {0.f,0.f,0.f,0.f,0.f,0.f,0.f,0.f};
#pragma unroll
    for (int i = 0; i < 4; ++i)
#pragma unroll
        for (int j = 0; j < 2; ++j) acc[i][j] = zero;

    const int KT = K / BK;
    stage(0, 0);
#if USE_ASYNC_LDS
    __builtin_amdgcn_s_wait_asynccnt(0);
#endif
    __syncthreads();

    const int arow = lane & 15;
    const int ah   = (lane >> 4) * 8;    // A: K half offset (elems)
    const int bk0  = (lane >> 4) * 8;    // B: K half offset (elems)

    for (int kt = 0; kt < KT; ++kt) {
        const int cur = kt & 1;
        if (kt + 1 < KT) stage(kt + 1, cur ^ 1);
        if (kt + 2 < KT) {  // gfx1250 global_prefetch_b8 two stages ahead
            __builtin_prefetch(aBase + (long)(kt + 2) * BK, 0, 1);
            __builtin_prefetch(bBase + (long)(kt + 2) * BK, 0, 1);
        }

#pragma unroll
        for (int s = 0; s < 2; ++s) {       // two WMMA K-steps per stage
            const int ko = s * 32;
            FragBH a[4], b[2];
#pragma unroll
            for (int i = 0; i < 4; ++i) {
                const bh* p = &sA[cur][wm + i * 16 + arow][ko];
                a[i].q[0] = *(const uint4*)(p + ah);
                a[i].q[1] = *(const uint4*)(p + ah + 16);
            }
#pragma unroll
            for (int j = 0; j < 2; ++j) {
                const bh* p = &sB[cur][wn + j * 16 + arow][ko];
                b[j].q[0] = *(const uint4*)(p + bk0);
                b[j].q[1] = *(const uint4*)(p + bk0 + 16);
            }
#pragma unroll
            for (int i = 0; i < 4; ++i)
#pragma unroll
                for (int j = 0; j < 2; ++j)
                    acc[i][j] = __builtin_amdgcn_wmma_f32_16x16x32_bf16(
                        false, a[i].v, false, b[j].v, (short)0, acc[i][j], false, false);
        }
#if USE_ASYNC_LDS
        __builtin_amdgcn_s_wait_asynccnt(0);
#endif
        __syncthreads();
    }
}

// ---- epilogues -------------------------------------------------------------

__device__ __forceinline__ void store_direct_f32(const v8f (&acc)[4][2], float* out, int ldo,
                                                 int mBlk, int nBlk)
{
    const int lane = threadIdx.x & 31, w = threadIdx.x >> 5;
    const int wm = mBlk + (w >> 2) * 64, wn = nBlk + (w & 3) * 32;
#pragma unroll
    for (int i = 0; i < 4; ++i)
#pragma unroll
        for (int j = 0; j < 2; ++j) {
            const int m0 = wm + i * 16 + ((lane >> 4) << 3);
            const int n  = wn + j * 16 + (lane & 15);
#pragma unroll
            for (int r = 0; r < 8; ++r)
                out[(long)(m0 + r) * ldo + n] = acc[i][j][r];
        }
}

__device__ __forceinline__ void store_direct_bf16(const v8f (&acc)[4][2], bh* out, int ldo,
                                                  int mBlk, int nBlk, const float* bias)
{
    const int lane = threadIdx.x & 31, w = threadIdx.x >> 5;
    const int wm = mBlk + (w >> 2) * 64, wn = nBlk + (w & 3) * 32;
#pragma unroll
    for (int i = 0; i < 4; ++i)
#pragma unroll
        for (int j = 0; j < 2; ++j) {
            const int m0 = wm + i * 16 + ((lane >> 4) << 3);
            const int n  = wn + j * 16 + (lane & 15);
#pragma unroll
            for (int r = 0; r < 8; ++r) {
                const float bb = bias ? bias[m0 + r] : 0.f;
                out[(long)(m0 + r) * ldo + n] = (bh)(acc[i][j][r] + bb);
            }
        }
}

// transposed bf16 store: out[n][m] (16B contiguous store per tile per lane)
__device__ __forceinline__ void store_trans_bf16(const v8f (&acc)[4][2], bh* out, int ldo,
                                                 int mBlk, int nBlk, const float* bias, float scale)
{
    const int lane = threadIdx.x & 31, w = threadIdx.x >> 5;
    const int wm = mBlk + (w >> 2) * 64, wn = nBlk + (w & 3) * 32;
#pragma unroll
    for (int i = 0; i < 4; ++i)
#pragma unroll
        for (int j = 0; j < 2; ++j) {
            const int m0 = wm + i * 16 + ((lane >> 4) << 3);
            const int n  = wn + j * 16 + (lane & 15);
            union { uint4 q; bh e[8]; } pk;
#pragma unroll
            for (int r = 0; r < 8; ++r)
                pk.e[r] = (bh)((acc[i][j][r] + bias[m0 + r]) * scale);
            *(uint4*)(out + (long)n * ldo + m0) = pk.q;
        }
}

// transposed f32 store with residual + per-col bias: out[n][m] = acc + bias[n] + resid[n][m]
__device__ __forceinline__ void store_trans_f32_resid(const v8f (&acc)[4][2], float* out, int ldo,
                                                      int mBlk, int nBlk, const float* bias,
                                                      const float* resid)
{
    const int lane = threadIdx.x & 31, w = threadIdx.x >> 5;
    const int wm = mBlk + (w >> 2) * 64, wn = nBlk + (w & 3) * 32;
#pragma unroll
    for (int i = 0; i < 4; ++i)
#pragma unroll
        for (int j = 0; j < 2; ++j) {
            const int m0 = wm + i * 16 + ((lane >> 4) << 3);
            const int n  = wn + j * 16 + (lane & 15);
            const float bo = bias[n];
            const float* xr = resid + (long)n * ldo + m0;
            float* dst      = out   + (long)n * ldo + m0;
            union { float4 q[2]; float e[8]; } rx, ro;
            rx.q[0] = ((const float4*)xr)[0];
            rx.q[1] = ((const float4*)xr)[1];
#pragma unroll
            for (int r = 0; r < 8; ++r)
                ro.e[r] = acc[i][j][r] + bo + rx.e[r];
            ((float4*)dst)[0] = ro.q[0];
            ((float4*)dst)[1] = ro.q[1];
        }
}

// ---------------------------------------------------------------------------
// K1: GroupNorm -> hT [B][N][C] bf16 (transposed so it is the Bt operand)
// One block per (batch, group): 16 channels x 2048 positions.
// ---------------------------------------------------------------------------
__global__ __launch_bounds__(256)
void groupnorm_kernel(const float* __restrict__ x, const float* __restrict__ gamma,
                      const float* __restrict__ beta, bh* __restrict__ hT)
{
    const int b = blockIdx.x >> 5, g = blockIdx.x & 31, tid = threadIdx.x;
    const float* xp = x + ((long)b * CC + (long)g * 16) * NN;

    float s = 0.f, s2 = 0.f;
    for (int i = tid; i < 16 * NN; i += 256) { const float v = xp[i]; s += v; s2 += v * v; }

    __shared__ float r0[256], r1[256];
    r0[tid] = s; r1[tid] = s2;
    __syncthreads();
    for (int off = 128; off > 0; off >>= 1) {
        if (tid < off) { r0[tid] += r0[tid + off]; r1[tid] += r1[tid + off]; }
        __syncthreads();
    }
    const float mean = r0[0] * (1.f / (16.f * NN));
    const float var  = r1[0] * (1.f / (16.f * NN)) - mean * mean;
    const float inv  = rsqrtf(var + EPSF);

    __shared__ float sa[16], sb[16];
    if (tid < 16) {
        const float gm = gamma[g * 16 + tid] * inv;
        sa[tid] = gm;
        sb[tid] = beta[g * 16 + tid] - mean * gm;
    }
    __syncthreads();

    __shared__ float stagef[16][257];
    bh* out = hT + (long)b * NN * CC + g * 16;
    for (int n0 = 0; n0 < NN; n0 += 256) {
#pragma unroll
        for (int c = 0; c < 16; ++c) stagef[c][tid] = xp[(long)c * NN + n0 + tid];
        __syncthreads();
        union { uint4 q[2]; bh e[16]; } pk;
#pragma unroll
        for (int c = 0; c < 16; ++c) pk.e[c] = (bh)(stagef[c][tid] * sa[c] + sb[c]);
        uint4* dst = (uint4*)(out + (long)(n0 + tid) * CC);
        dst[0] = pk.q[0]; dst[1] = pk.q[1];
        __syncthreads();
    }
}

// K2: f32 -> bf16 weight conversion (wq/wk/wv row-major = A ops; wp row-major = Bt op)
__global__ void convert_w(const float* __restrict__ wq, const float* __restrict__ wk,
                          const float* __restrict__ wv, const float* __restrict__ wp,
                          bh* __restrict__ oq, bh* __restrict__ ok,
                          bh* __restrict__ ov, bh* __restrict__ op)
{
    const int i = blockIdx.x * 256 + threadIdx.x;
    if (i < CC * CC) {
        oq[i] = (bh)wq[i]; ok[i] = (bh)wk[i];
        ov[i] = (bh)wv[i]; op[i] = (bh)wp[i];
    }
}

// K3: q/k/v = W x h  (per batch, per weight). q,k stored transposed [N][C] (q pre-scaled),
// v stored [C][N] (so it is the Bt operand of the PV GEMM).
__global__ __launch_bounds__(256)
void gemm_qkv(const bh* __restrict__ wqb, const bh* __restrict__ wkb, const bh* __restrict__ wvb,
              const bh* __restrict__ hT,
              const float* __restrict__ bq, const float* __restrict__ bk, const float* __restrict__ bv,
              bh* __restrict__ qT, bh* __restrict__ kT, bh* __restrict__ vmat)
{
    const int z = blockIdx.z, b = z / 3, w = z % 3;
    const bh* A  = (w == 0) ? wqb : ((w == 1) ? wkb : wvb);
    const bh* Bt = hT + (long)b * NN * CC;
    const int mBlk = blockIdx.y * BM;   // over o (512)
    const int nBlk = blockIdx.x * BN;   // over n (2048)
    v8f acc[4][2];
    gemm_core(A, CC, Bt, CC, CC, mBlk, nBlk, acc);
    if (w == 0)
        store_trans_bf16(acc, qT + (long)b * NN * CC, CC, mBlk, nBlk, bq, QSCALE);
    else if (w == 1)
        store_trans_bf16(acc, kT + (long)b * NN * CC, CC, mBlk, nBlk, bk, 1.f);
    else
        store_direct_bf16(acc, vmat + (long)b * CC * NN, NN, mBlk, nBlk, bv);
}

// K4: S[n][m] = sum_c qT[n][c] * kT[m][c]   (f32 logits, scale already in q)
__global__ __launch_bounds__(256)
void gemm_attn(const bh* __restrict__ qT, const bh* __restrict__ kT, float* __restrict__ S)
{
    const int b = blockIdx.z;
    const int mBlk = blockIdx.y * BM, nBlk = blockIdx.x * BN;
    v8f acc[4][2];
    gemm_core(qT + (long)b * NN * CC, CC, kT + (long)b * NN * CC, CC, CC, mBlk, nBlk, acc);
    store_direct_f32(acc, S + (long)b * NN * NN, NN, mBlk, nBlk);
}

// K5: row softmax over m; P (bf16) packed into the front half of the same S row.
__global__ __launch_bounds__(256)
void softmax_kernel(float* __restrict__ S)
{
    float* sp = S + (long)blockIdx.x * NN;
    bh*   pp  = (bh*)sp;
    const int tid = threadIdx.x;

    float v[8];
#pragma unroll
    for (int i = 0; i < 8; ++i) v[i] = sp[tid + i * 256];

    float mx = v[0];
#pragma unroll
    for (int i = 1; i < 8; ++i) mx = fmaxf(mx, v[i]);

    __shared__ float red[256];
    red[tid] = mx; __syncthreads();
    for (int off = 128; off > 0; off >>= 1) {
        if (tid < off) red[tid] = fmaxf(red[tid], red[tid + off]);
        __syncthreads();
    }
    mx = red[0];
    __syncthreads();

    float e[8], s = 0.f;
#pragma unroll
    for (int i = 0; i < 8; ++i) { e[i] = __expf(v[i] - mx); s += e[i]; }
    red[tid] = s; __syncthreads();
    for (int off = 128; off > 0; off >>= 1) {
        if (tid < off) red[tid] += red[tid + off];
        __syncthreads();
    }
    const float inv = 1.f / red[0];
    __syncthreads();   // all reads of sp done before in-place bf16 writes

#pragma unroll
    for (int i = 0; i < 8; ++i) pp[tid + i * 256] = (bh)(e[i] * inv);
}

// K6: oT[n][c] = sum_m P[n][m] * v[c][m]   (P row stride = 2*NN bf16 elems)
__global__ __launch_bounds__(256)
void gemm_pv(const float* __restrict__ S, const bh* __restrict__ vmat, bh* __restrict__ oT)
{
    const int b = blockIdx.z;
    const bh* A  = (const bh*)(S + (long)b * NN * NN);   // lda = 2*NN
    const bh* Bt = vmat + (long)b * CC * NN;             // ldb = NN (rows = c)
    const int mBlk = blockIdx.y * BM;   // over n
    const int nBlk = blockIdx.x * BN;   // over c
    v8f acc[4][2];
    gemm_core(A, 2 * NN, Bt, NN, NN, mBlk, nBlk, acc);
    store_direct_bf16(acc, oT + (long)b * NN * CC, CC, mBlk, nBlk, nullptr);
}

// K7: out[b][o][n] = x[b][o][n] + bp[o] + sum_c oT[n][c] * wp[o][c]
__global__ __launch_bounds__(256)
void gemm_proj(const bh* __restrict__ oT, const bh* __restrict__ wpb,
               const float* __restrict__ bp, const float* __restrict__ x,
               float* __restrict__ out)
{
    const int b = blockIdx.z;
    const int mBlk = blockIdx.y * BM;   // over n
    const int nBlk = blockIdx.x * BN;   // over o
    v8f acc[4][2];
    gemm_core(oT + (long)b * NN * CC, CC, wpb, CC, CC, mBlk, nBlk, acc);
    store_trans_f32_resid(acc, out + (long)b * CC * NN, NN, mBlk, nBlk,
                          bp, x + (long)b * CC * NN);
}

// ---------------------------------------------------------------------------
extern "C" void kernel_launch(void* const* d_in, const int* in_sizes, int n_in,
                              void* d_out, int out_size, void* d_ws, size_t ws_size,
                              hipStream_t stream)
{
    const float* x   = (const float*)d_in[0];
    const float* gns = (const float*)d_in[1];
    const float* gnb = (const float*)d_in[2];
    const float* wq  = (const float*)d_in[3];
    const float* bq  = (const float*)d_in[4];
    const float* wk  = (const float*)d_in[5];
    const float* bk  = (const float*)d_in[6];
    const float* wv  = (const float*)d_in[7];
    const float* bv  = (const float*)d_in[8];
    const float* wp  = (const float*)d_in[9];
    const float* bp  = (const float*)d_in[10];
    float* out = (float*)d_out;

    char* ws = (char*)d_ws;
    size_t off = 0;
    auto walloc = [&](size_t bytes) -> void* {
        void* p = ws + off;
        off = (off + bytes + 255) & ~(size_t)255;
        return p;
    };
    const size_t planeBH = (size_t)BBATCH * NN * CC * sizeof(bh);
    bh* hT  = (bh*)walloc(planeBH);
    bh* qT  = (bh*)walloc(planeBH);
    bh* kT  = (bh*)walloc(planeBH);
    bh* vmt = (bh*)walloc(planeBH);
    bh* oT  = (bh*)walloc(planeBH);
    bh* wqb = (bh*)walloc((size_t)CC * CC * sizeof(bh));
    bh* wkb = (bh*)walloc((size_t)CC * CC * sizeof(bh));
    bh* wvb = (bh*)walloc((size_t)CC * CC * sizeof(bh));
    bh* wpb = (bh*)walloc((size_t)CC * CC * sizeof(bh));
    float* S = (float*)walloc((size_t)BBATCH * NN * NN * sizeof(float));

    groupnorm_kernel<<<BBATCH * GG, 256, 0, stream>>>(x, gns, gnb, hT);
    convert_w<<<(CC * CC + 255) / 256, 256, 0, stream>>>(wq, wk, wv, wp, wqb, wkb, wvb, wpb);
    gemm_qkv<<<dim3(NN / BN, CC / BM, BBATCH * 3), 256, 0, stream>>>(
        wqb, wkb, wvb, hT, bq, bk, bv, qT, kT, vmt);
    gemm_attn<<<dim3(NN / BN, NN / BM, BBATCH), 256, 0, stream>>>(qT, kT, S);
    softmax_kernel<<<BBATCH * NN, 256, 0, stream>>>(S);
    gemm_pv<<<dim3(CC / BN, NN / BM, BBATCH), 256, 0, stream>>>(S, vmt, oT);
    gemm_proj<<<dim3(CC / BN, NN / BM, BBATCH), 256, 0, stream>>>(oT, wpb, bp, x, out);
}
